// KGMC_autoencoder_77919296684696
// MI455X (gfx1250) — compile-verified, matrix operand order checked by reading, not run on hardware
//
#include <hip/hip_runtime.h>
#include <hip/hip_bf16.h>

#define NNODES 65536
#define LAT 32
#define NCOLS 288          // 8 relations * 32 + self-loop 32

typedef __attribute__((ext_vector_type(16))) __bf16 v16bf;
typedef __attribute__((ext_vector_type(8)))  float  v8f;
typedef __attribute__((ext_vector_type(4)))  unsigned int u32x4;

// ---------- fp32 -> bf16 conversion ----------
__global__ void k_tobf16(const float* __restrict__ x, __bf16* __restrict__ xb, long n) {
    long t = (long)blockIdx.x * blockDim.x + threadIdx.x;
    if (t < n) xb[t] = (__bf16)x[t];
}

// ---------- zero fill ----------
__global__ void k_zero(float* __restrict__ p, long n) {
    long t = (long)blockIdx.x * blockDim.x + threadIdx.x;
    if (t < n) p[t] = 0.f;
}

// ---------- pack [R][K][32] weights + [K][32] self-loop into Wt[288][K] bf16 ----------
__global__ void k_pack(const float* __restrict__ W, const float* __restrict__ Lw,
                       __bf16* __restrict__ out, int K) {
    int t = blockIdx.x * blockDim.x + threadIdx.x;
    if (t >= NCOLS * K) return;
    int c = t / K, k = t - c * K;
    float v;
    if (c < 256) {
        int r = c >> 5, col = c & 31;
        v = W[(size_t)r * K * 32 + (size_t)k * 32 + col];
    } else {
        v = Lw[(size_t)k * 32 + (c - 256)];
    }
    out[(size_t)c * K + k] = (__bf16)v;
}

// ---------- dense GEMM: Y[N,288] = Xb[N,K] @ Wt^T  via v_wmma_f32_16x16x32_bf16 ----------
// One wave per 16x16 output tile. A: 16-bit A-matrix layout (lane<16: M=lane,
// K chunks [k0..k0+7][k0+16..k0+23]; lane>=16: K chunks +8/+24). B symmetric
// (lane carries output column, Wt rows are weight columns => contiguous loads).
__global__ void k_gemm_wmma(const __bf16* __restrict__ Xb, const __bf16* __restrict__ Wt,
                            float* __restrict__ Y, int K) {
    int wave = threadIdx.x >> 5;
    int lane = threadIdx.x & 31;
    int half = lane >> 4;
    int mr   = lane & 15;
    int m0 = (blockIdx.x * 8 + wave) * 16;
    int n0 = blockIdx.y * 16;

    v8f c = {};
    for (int k0 = 0; k0 < K; k0 += 32) {
        union { v16bf v; u32x4 u[2]; } a, b;
        const __bf16* pa = Xb + (size_t)(m0 + mr) * K + k0 + half * 8;
        const __bf16* pb = Wt + (size_t)(n0 + mr) * K + k0 + half * 8;
        a.u[0] = *(const u32x4*)(pa);
        a.u[1] = *(const u32x4*)(pa + 16);
        b.u[0] = *(const u32x4*)(pb);
        b.u[1] = *(const u32x4*)(pb + 16);
        c = __builtin_amdgcn_wmma_f32_16x16x32_bf16(false, a.v, false, b.v,
                                                    (short)0, c, false, false);
    }
#pragma unroll
    for (int g = 0; g < 8; ++g) {
        int row = m0 + half * 8 + g;          // C/D layout: vgpr g -> M = g + 8*half
        Y[(size_t)row * NCOLS + n0 + mr] = c[g];
    }
}

// ---------- edge scatter: acc[dst] += Y[src, etype*32 + j] (one lane per column) ----------
__global__ void k_scatter(const int* __restrict__ src, const int* __restrict__ dst,
                          const int* __restrict__ et, const float* __restrict__ Y,
                          float* __restrict__ acc, long nE) {
    long tid = (long)blockIdx.x * blockDim.x + threadIdx.x;
    long e = tid >> 5;
    int  j = (int)(tid & 31);
    if (e >= nE) return;
    int s = src[e], d = dst[e], r = et[e];
    atomicAdd(acc + (size_t)d * 32 + j, Y[(size_t)s * NCOLS + r * 32 + j]);
}

// ---------- combine: h = tanh(acc + Y[:,rel0] + Y[:,self] + b); write cs col-block + bf16 copy ----------
__global__ void k_combine(const float* __restrict__ acc, const float* __restrict__ Y,
                          const float* __restrict__ b, float* __restrict__ cs,
                          __bf16* __restrict__ xb, int layer, int n) {
    long tid = (long)blockIdx.x * blockDim.x + threadIdx.x;
    int i = (int)(tid >> 5), j = (int)(tid & 31);
    if (i >= n) return;
    float v = acc[(size_t)i * 32 + j]
            + Y[(size_t)i * NCOLS + j]            // self-loop edge (etype 0)
            + Y[(size_t)i * NCOLS + 256 + j]      // feat @ Lw
            + b[j];
    float t = tanhf(v);
    cs[(size_t)i * 96 + layer * 32 + j] = t;
    xb[(size_t)i * 32 + j] = (__bf16)t;           // next layer's GEMM input (K=32)
}

// ---------- VGAE head: z = (res@Wmu+bmu) + noise * exp(res@Wls+bls); one wave per row ----------
__global__ void k_vgae(const float* __restrict__ cs, const int* __restrict__ uidx,
                       const int* __restrict__ iidx, const float* __restrict__ noise,
                       const float* __restrict__ Wmu, const float* __restrict__ bmu,
                       const float* __restrict__ Wls, const float* __restrict__ bls,
                       float* __restrict__ z, int nh) {
    __shared__ float sMu[96 * 32], sLs[96 * 32], sbm[32], sbl[32];
    for (int t = threadIdx.x; t < 96 * 32; t += 256) { sMu[t] = Wmu[t]; sLs[t] = Wls[t]; }
    if (threadIdx.x < 32) { sbm[threadIdx.x] = bmu[threadIdx.x]; sbl[threadIdx.x] = bls[threadIdx.x]; }
    __syncthreads();
    int wave = threadIdx.x >> 5, j = threadIdx.x & 31;
    int i = blockIdx.x * 8 + wave;
    int idx = (i < nh) ? uidx[i] : iidx[i - nh];
    const float* row = cs + (size_t)idx * 96;
    float mu = 0.f, ls = 0.f;
#pragma unroll 4
    for (int k = 0; k < 96; ++k) {
        float f = row[k];
        mu += f * sMu[k * 32 + j];
        ls += f * sLs[k * 32 + j];
    }
    mu += sbm[j]; ls += sbl[j];
    z[(size_t)i * 32 + j] = mu + noise[(size_t)i * 32 + j] * expf(ls);
}

// ---------- decoder MLP: sigmoid(relu([z_u,z_i]@Wl1+b)@Wl2+b2); one wave per row ----------
__global__ void k_mlp(const float* __restrict__ z, const float* __restrict__ Wl1,
                      const float* __restrict__ bl1, const float* __restrict__ Wl2,
                      const float* __restrict__ bl2, float* __restrict__ out, int nh) {
    __shared__ float sW[64 * 64], sb[64], sW2[64], sb2;
    for (int t = threadIdx.x; t < 64 * 64; t += 256) sW[t] = Wl1[t];
    if (threadIdx.x < 64) { sb[threadIdx.x] = bl1[threadIdx.x]; sW2[threadIdx.x] = Wl2[threadIdx.x]; }
    if (threadIdx.x == 0) sb2 = bl2[0];
    __syncthreads();
    int wave = threadIdx.x >> 5, j = threadIdx.x & 31;
    int i = blockIdx.x * 8 + wave;
    if (i >= nh) return;
    float a1 = sb[j], a2 = sb[j + 32];
    const float* z0 = z + (size_t)i * 32;
    const float* z1 = z + (size_t)(i + nh) * 32;
#pragma unroll 4
    for (int k = 0; k < 32; ++k) {
        float h = z0[k];
        a1 += h * sW[k * 64 + j];
        a2 += h * sW[k * 64 + j + 32];
    }
#pragma unroll 4
    for (int k = 0; k < 32; ++k) {
        float h = z1[k];
        a1 += h * sW[(k + 32) * 64 + j];
        a2 += h * sW[(k + 32) * 64 + j + 32];
    }
    a1 = fmaxf(a1, 0.f); a2 = fmaxf(a2, 0.f);
    float p = a1 * sW2[j] + a2 * sW2[j + 32];
#pragma unroll
    for (int off = 16; off > 0; off >>= 1) p += __shfl_down(p, off, 32);
    if (j == 0) out[i] = 1.f / (1.f + expf(-(p + sb2)));
}

extern "C" void kernel_launch(void* const* d_in, const int* in_sizes, int n_in,
                              void* d_out, int out_size, void* d_ws, size_t ws_size,
                              hipStream_t stream) {
    const float* x     = (const float*)d_in[0];
    const int*   src   = (const int*)d_in[1];
    const int*   dst   = (const int*)d_in[2];
    const int*   etype = (const int*)d_in[3];
    const int*   uidx  = (const int*)d_in[4];
    const int*   iidx  = (const int*)d_in[5];
    const float* noise = (const float*)d_in[6];
    const float* W0 = (const float*)d_in[7];  const float* b0 = (const float*)d_in[8];
    const float* L0 = (const float*)d_in[9];
    const float* W1 = (const float*)d_in[10]; const float* b1 = (const float*)d_in[11];
    const float* L1 = (const float*)d_in[12];
    const float* W2 = (const float*)d_in[13]; const float* b2 = (const float*)d_in[14];
    const float* L2 = (const float*)d_in[15];
    const float* Wmu = (const float*)d_in[16]; const float* bmu = (const float*)d_in[17];
    const float* Wls = (const float*)d_in[18]; const float* bls = (const float*)d_in[19];
    const float* Wl1 = (const float*)d_in[20]; const float* bl1 = (const float*)d_in[21];
    const float* Wl2 = (const float*)d_in[22]; const float* bl2 = (const float*)d_in[23];

    const int  N  = NNODES;
    const long E  = in_sizes[1];
    const int  NH = in_sizes[4];   // N/2

    // workspace carve-up (all offsets multiples of 256B)
    char* w = (char*)d_ws;
    float*  cs  = (float*)w;  w += (size_t)N * 96 * 4;      // concat h0|h1|h2
    float*  Y   = (float*)w;  w += (size_t)N * NCOLS * 4;   // per-relation transforms (L2-resident)
    float*  acc = (float*)w;  w += (size_t)N * 32 * 4;      // scatter accumulator
    float*  zb  = (float*)w;  w += (size_t)N * 32 * 4;      // latent z
    __bf16* xb  = (__bf16*)w; w += (size_t)N * 64 * 2;      // bf16 GEMM input
    __bf16* wt0 = (__bf16*)w; w += (size_t)NCOLS * 64 * 2;
    __bf16* wt1 = (__bf16*)w; w += (size_t)NCOLS * 32 * 2;
    __bf16* wt2 = (__bf16*)w; w += (size_t)NCOLS * 32 * 2;

    // pack weights (fp32 -> transposed bf16 fragments)
    k_pack<<<(NCOLS * 64 + 255) / 256, 256, 0, stream>>>(W0, L0, wt0, 64);
    k_pack<<<(NCOLS * 32 + 255) / 256, 256, 0, stream>>>(W1, L1, wt1, 32);
    k_pack<<<(NCOLS * 32 + 255) / 256, 256, 0, stream>>>(W2, L2, wt2, 32);
    k_tobf16<<<(int)(((long)N * 64 + 255) / 256), 256, 0, stream>>>(x, xb, (long)N * 64);

    const __bf16* wts[3] = {wt0, wt1, wt2};
    const float*  bs[3]  = {b0, b1, b2};
    const int     Ks[3]  = {64, 32, 32};

    for (int l = 0; l < 3; ++l) {
        k_zero<<<(int)(((long)N * 32 + 255) / 256), 256, 0, stream>>>(acc, (long)N * 32);
        dim3 grid(N / 16 / 8, NCOLS / 16);                  // 512 x 18, 8 waves/block
        k_gemm_wmma<<<grid, 256, 0, stream>>>(xb, wts[l], Y, Ks[l]);
        k_scatter<<<(int)((E * 32 + 255) / 256), 256, 0, stream>>>(src, dst, etype, Y, acc, E);
        k_combine<<<(int)(((long)N * 32 + 255) / 256), 256, 0, stream>>>(acc, Y, bs[l], cs, xb, l, N);
    }

    k_vgae<<<N / 8, 256, 0, stream>>>(cs, uidx, iidx, noise, Wmu, bmu, Wls, bls, zb, NH);
    k_mlp<<<(NH + 7) / 8, 256, 0, stream>>>(zb, Wl1, bl1, Wl2, bl2, (float*)d_out, NH);
}
